// Attention_56530359550576
// MI455X (gfx1250) — matrix operand order repaired and checked by reading, not compiled, not run on hardware
//
#include <hip/hip_runtime.h>

#define B_   8
#define LQ_  2048
#define LK_  2048
#define DIM_ 128

typedef __attribute__((ext_vector_type(16))) __bf16 v16bf;
typedef __attribute__((ext_vector_type(8)))  __bf16 v8bf;
typedef __attribute__((ext_vector_type(8)))  float  v8f;

// ---------------------------------------------------------------------------
// Kernel 1: one-time conversion of context (f32) into
//   Kb[b][k][d]  : bf16, row-major   (B-operand for Q.K^T)
//   Vt[b][d][k]  : bf16, transposed  (B-operand for P.V)
// LDS-tiled transpose so both reads and writes are coalesced.
// ---------------------------------------------------------------------------
__global__ void __launch_bounds__(256) cvt_ctx_kernel(const float* __restrict__ ctx,
                                                      __bf16* __restrict__ Kb,
                                                      __bf16* __restrict__ Vt) {
  __shared__ float tile[32][33];
  const int tx = threadIdx.x;      // 0..31
  const int ty = threadIdx.y;      // 0..7
  const int d0 = blockIdx.x * 32;
  const int k0 = blockIdx.y * 32;
  const int b  = blockIdx.z;
#pragma unroll
  for (int r = 0; r < 4; ++r) {
    const int kk = ty + 8 * r;
    const size_t idx = ((size_t)b * LK_ + (size_t)(k0 + kk)) * DIM_ + d0 + tx;
    const float v = ctx[idx];
    tile[kk][tx] = v;
    Kb[idx] = (__bf16)v;
  }
  __syncthreads();
#pragma unroll
  for (int r = 0; r < 4; ++r) {
    const int dd = ty + 8 * r;
    Vt[((size_t)b * DIM_ + (size_t)(d0 + dd)) * LK_ + k0 + tx] = (__bf16)tile[tx][dd];
  }
}

// ---------------------------------------------------------------------------
// Main kernel: 1 wave (32 threads) = one 16-row Q tile over all 2048 keys.
// Pass 1: scores via WMMA (once), mask, write raw logits to out_attn,
//         track per-lane row max.
// Pass 2: read logits back (L2), p = exp(s-M), accumulate L and O += P.V
//         via WMMA, routing P through a padded LDS tile (C-layout->A-layout).
// Pass 3: out -> normalized; out_attn -= (M + log L) in a coalesced sweep.
// ---------------------------------------------------------------------------
__global__ void __launch_bounds__(32) attn_kernel(const float* __restrict__ Q,
                                                  const unsigned char* __restrict__ mask,
                                                  const __bf16* __restrict__ Kb,
                                                  const __bf16* __restrict__ Vt,
                                                  float* __restrict__ out,
                                                  float* __restrict__ out_attn) {
  // P tile staging: 16 rows x 32 cols bf16, row stride 40 elems (80B: 16B
  // aligned, bank base r*20 mod 64 distinct for r=0..15 -> conflict-free)
  __shared__ __align__(16) __bf16 plds[16 * 40];
  __shared__ float llds[16];

  const int lane = threadIdx.x & 31;
  const int n = lane & 15;
  const int h = lane >> 4;
  const int qt = blockIdx.x;               // 0..1023
  const int b  = qt >> 7;                  // LQ_/16 == 128 tiles per batch
  const int q0 = (qt & 127) << 4;
  const float scale = 0.08838834764831845f;  // 1/sqrt(DIM), folded into Q

  // ---- Build Q A-fragments (16x32 bf16 per 32-wide D chunk), contiguous k order:
  // lane row = n, element e -> K = 16h+e
  v16bf aq[4];
  const float* qrow = Q + ((size_t)b * LQ_ + (size_t)(q0 + n)) * DIM_;
#pragma unroll
  for (int c = 0; c < 4; ++c) {
#pragma unroll
    for (int e = 0; e < 8; ++e) {
      aq[c][e]     = (__bf16)(qrow[32 * c + 16 * h + e] * scale);
      aq[c][e + 8] = (__bf16)(qrow[32 * c + 16 * h + 8 + e] * scale);
    }
  }

  // ---- Pass 1: one WMMA sweep over all 128 K tiles
  float M[8];
#pragma unroll
  for (int v = 0; v < 8; ++v) M[v] = -__builtin_inff();

  const __bf16* kp0 = Kb + ((size_t)b * LK_ + n) * DIM_ + 16 * h;
  const unsigned char* mp0 =
      mask + ((size_t)b * LQ_ + (size_t)(q0 + 8 * h)) * LK_ + n;
  float* oa0 = out_attn + ((size_t)b * LQ_ + (size_t)(q0 + 8 * h)) * LK_ + n;

  for (int j = 0; j < 128; ++j) {
    const __bf16* kp = kp0 + (size_t)j * 16 * DIM_;
    __builtin_prefetch(kp + 16 * DIM_, 0, 3);  // next K tile -> global_prefetch
    v8f acc = {0.f, 0.f, 0.f, 0.f, 0.f, 0.f, 0.f, 0.f};
#pragma unroll
    for (int c = 0; c < 4; ++c) {
      v16bf bk = *(const v16bf*)(kp + 32 * c);  // B layout: k = e + 16h
      acc = __builtin_amdgcn_wmma_f32_16x16x32_bf16(false, aq[c], false, bk,
                                                    (short)0, acc, false, false);
    }
    const unsigned char* mp = mp0 + j * 16;
    float* oa = oa0 + j * 16;
#pragma unroll
    for (int v = 0; v < 8; ++v) {
      const float s = mp[(size_t)v * LK_] ? -__builtin_inff() : acc[v];
      oa[(size_t)v * LK_] = s;            // raw masked logit (fixed up later)
      M[v] = fmaxf(M[v], s);
    }
  }
#pragma unroll
  for (int v = 0; v < 8; ++v) {
#pragma unroll
    for (int off = 8; off >= 1; off >>= 1)
      M[v] = fmaxf(M[v], __shfl_xor(M[v], off, 32));
  }

  // ---- Pass 2: read logits back (L2-resident), accumulate L and O += P.V
  v8f O[8];
#pragma unroll
  for (int t = 0; t < 8; ++t) O[t] = (v8f){0.f, 0.f, 0.f, 0.f, 0.f, 0.f, 0.f, 0.f};
  float Lacc[8];
#pragma unroll
  for (int v = 0; v < 8; ++v) Lacc[v] = 0.f;

  for (int g = 0; g < 64; ++g) {
#pragma unroll
    for (int T = 0; T < 2; ++T) {
      const int j = 2 * g + T;
      const float* oa = oa0 + j * 16;
#pragma unroll
      for (int v = 0; v < 8; ++v) {
        const float s = oa[(size_t)v * LK_];   // same-wave store->load: in order
        const float p = __expf(s - M[v]);      // masked (-inf) -> 0
        Lacc[v] += p;
        plds[(v + 8 * h) * 40 + 16 * T + n] = (__bf16)p;
      }
    }
    // cross-lane LDS exchange within the wave; DS ops are in-order per wave,
    // make the dependency explicit anyway.
    asm volatile("s_wait_dscnt 0x0" ::: "memory");
    const __bf16* rowp = &plds[n * 40];
    const v8bf lo = *(const v8bf*)(rowp + 16 * h);
    const v8bf hi = *(const v8bf*)(rowp + 16 * h + 8);
    v16bf ap;
#pragma unroll
    for (int e = 0; e < 8; ++e) { ap[e] = lo[e]; ap[e + 8] = hi[e]; }
    // O[t] += P(16x32) . V(32x16) for each 16-wide D chunk, B from Vt rows
#pragma unroll
    for (int t = 0; t < 8; ++t) {
      const __bf16* vp = Vt + ((size_t)b * DIM_ + (size_t)(16 * t + n)) * LK_ +
                         32 * g + 16 * h;
      const v16bf bv = *(const v16bf*)vp;
      O[t] = __builtin_amdgcn_wmma_f32_16x16x32_bf16(false, ap, false, bv,
                                                     (short)0, O[t], false, false);
    }
  }

  // ---- Reduce L across the 16 lanes holding each row; fl = M + log L
#pragma unroll
  for (int v = 0; v < 8; ++v) {
#pragma unroll
    for (int off = 8; off >= 1; off >>= 1) Lacc[v] += __shfl_xor(Lacc[v], off, 32);
  }
  float fl[8];
#pragma unroll
  for (int v = 0; v < 8; ++v) fl[v] = M[v] + __logf(Lacc[v]);
  if (n == 0) {
#pragma unroll
    for (int v = 0; v < 8; ++v) llds[v + 8 * h] = fl[v];
  }

  // ---- Normalize and store `out`
#pragma unroll
  for (int t = 0; t < 8; ++t) {
#pragma unroll
    for (int v = 0; v < 8; ++v) {
      out[((size_t)b * LQ_ + (size_t)(q0 + v + 8 * h)) * DIM_ + 16 * t + n] =
          O[t][v] / Lacc[v];
    }
  }

  asm volatile("s_wait_dscnt 0x0" ::: "memory");

  // ---- Pass 3: out_attn = s - (M + log L), coalesced float4 sweep over this
  // wave's 16x2048 slab (dirty in L2, HBM still sees a single write)
  float4* oab = (float4*)(out_attn + ((size_t)b * LQ_ + (size_t)q0) * LK_);
  for (int i = 0; i < 256; ++i) {
    const int idx = i * 32 + lane;     // 512 float4 per row, uniform row per iter
    const int row = idx >> 9;
    const float ll = llds[row];
    float4 v = oab[idx];
    v.x -= ll; v.y -= ll; v.z -= ll; v.w -= ll;
    oab[idx] = v;
  }
}

// ---------------------------------------------------------------------------
extern "C" void kernel_launch(void* const* d_in, const int* in_sizes, int n_in,
                              void* d_out, int out_size, void* d_ws, size_t ws_size,
                              hipStream_t stream) {
  (void)in_sizes; (void)n_in; (void)out_size; (void)ws_size;
  const float* Q            = (const float*)d_in[0];          // "output" [B,LQ,D] f32
  const float* ctx          = (const float*)d_in[1];          // "context" [B,LK,D] f32
  const unsigned char* mask = (const unsigned char*)d_in[2];  // bool [B,LQ,LK]

  float* out      = (float*)d_out;                                  // [B,LQ,D]
  float* out_attn = out + (size_t)B_ * LQ_ * DIM_;                  // [B,LQ,LK]

  __bf16* Kb = (__bf16*)d_ws;                                       // 4 MB
  __bf16* Vt = Kb + (size_t)B_ * LK_ * DIM_;                        // 4 MB

  dim3 cb(32, 8);
  dim3 cg(DIM_ / 32, LK_ / 32, B_);
  cvt_ctx_kernel<<<cg, cb, 0, stream>>>(ctx, Kb, Vt);

  const int blocks = B_ * (LQ_ / 16);  // 1024 single-wave workgroups
  attn_kernel<<<blocks, 32, 0, stream>>>(Q, mask, Kb, Vt, out, out_attn);
}
